// Transformer_85899346010
// MI455X (gfx1250) — matrix operand order repaired and checked by reading, compiled, not run on hardware
//
#include <hip/hip_runtime.h>
#include <math.h>

// ---------------- problem constants ----------------
constexpr int cB  = 4;
constexpr int cL  = 2048;
constexpr int cD  = 512;
constexpr int cH  = 8;
constexpr int cHD = 64;        // cD / cH
constexpr int cFF = 2048;
constexpr int cNL = 3;
constexpr int c3D = 3 * cD;    // 1536
constexpr int cBL = cB * cL;   // 8192
constexpr float cSCALE = 0.044194173824159216f; // 512^-0.5
constexpr float cEPS = 1e-5f;

typedef __attribute__((ext_vector_type(16))) _Float16 v16h;
typedef __attribute__((ext_vector_type(8)))  _Float16 v8h;
typedef __attribute__((ext_vector_type(8)))  float    v8f;

// ---------------- WMMA helpers ----------------
static __device__ __forceinline__ v8f wmma_f16(v16h a, v16h b, v8f c) {
  // D = A(16x32 f16) * B(32x16 f16) + C(16x16 f32)
  return __builtin_amdgcn_wmma_f32_16x16x32_f16(
      /*neg_a=*/false, a, /*neg_b=*/false, b,
      /*c_mod=*/(short)0, c, /*reuse_a=*/false, /*reuse_b=*/false);
}

static __device__ __forceinline__ v16h frag_from_2x8(const _Float16* p0,
                                                     const _Float16* p1) {
  v8h a = *(const v8h*)p0;
  v8h b = *(const v8h*)p1;
  v16h r;
#pragma unroll
  for (int e = 0; e < 8; ++e) { r[e] = a[e]; r[e + 8] = b[e]; }
  return r;
}

// A fragment (16x32 f16), A row-major, caller pre-offsets to (mbase,kbase).
// Lane: m = lane&15, half = lane>>4; K = half*8+e (e<8), 16+half*8+(e-8).
static __device__ __forceinline__ v16h load_a(const _Float16* A, int lda, int lane) {
  int m = lane & 15, hf = lane >> 4;
  const _Float16* p = A + (size_t)m * lda + hf * 8;
  return frag_from_2x8(p, p + 16);
}

// B fragment (32x16 f16) from Bt where B[k][n] == Bt[n*ldb + k].
// Lane: n = lane&15; K = (lane>>4)*8+e (e<8), 16+(lane>>4)*8+(e-8).
static __device__ __forceinline__ v16h load_b(const _Float16* Bt, int ldb, int lane) {
  int n = lane & 15, g = lane >> 4;
  const _Float16* p = Bt + (size_t)n * ldb + g * 8;
  return frag_from_2x8(p, p + 16);
}

// ---------------- prep kernels ----------------
__global__ void add_pos_kernel(const float* __restrict__ x, const float* __restrict__ pos,
                               float* __restrict__ xf, _Float16* __restrict__ xh) {
  size_t i = (size_t)blockIdx.x * blockDim.x + threadIdx.x;
  if (i >= (size_t)cBL * cD) return;
  float v = x[i] + pos[i % ((size_t)cL * cD)];
  xf[i] = v;
  xh[i] = (_Float16)v;
}

__global__ void log_mask_kernel(const float* __restrict__ mask, float* __restrict__ lm) {
  size_t i = (size_t)blockIdx.x * blockDim.x + threadIdx.x;
  if (i >= (size_t)cBL) return;
  lm[i] = __logf(mask[i]);
}

// W[K][N] f32 -> Wt[N][K] f16
__global__ void conv_wt_kernel(const float* __restrict__ W, _Float16* __restrict__ Wt,
                               int K, int N) {
  size_t i = (size_t)blockIdx.x * blockDim.x + threadIdx.x;
  if (i >= (size_t)K * N) return;
  int n = (int)(i % N);
  int k = (int)(i / N);
  Wt[(size_t)n * K + k] = (_Float16)W[i];
}

// V slice of qkv -> Vt[b][h][hd][l]
__global__ void transpose_v_kernel(const _Float16* __restrict__ qkvh,
                                   _Float16* __restrict__ vt) {
  size_t i = (size_t)blockIdx.x * blockDim.x + threadIdx.x;
  if (i >= (size_t)cB * cH * cHD * cL) return;
  int l = (int)(i & (cL - 1));
  size_t r = i >> 11;            // / cL
  int hd = (int)(r & (cHD - 1));
  r >>= 6;                       // / cHD
  int h = (int)(r & (cH - 1));
  int b = (int)(r >> 3);         // / cH
  vt[i] = qkvh[((size_t)b * cL + l) * c3D + h * (3 * cHD) + 2 * cHD + hd];
}

// ---------------- WMMA GEMM: C[M,N] = A[M,K] @ W[K,N] + bias ----------------
// A f16 row-major, Wt f16 = W transposed [N][K]. K must be a multiple of 64.
// Wave tile 32x64 (8 WMMA tiles); block tile 128x128 (4 waves on M, 2 on N).
// Software-pipelined: ping/pong fragment double buffering so global loads
// overlap WMMA issue instead of s_wait_loadcnt 0x0 per step.
// mode 0: f16 out, scale q columns (qkv). mode 1: f32 out. mode 2: f16 out + relu.
__global__ __launch_bounds__(256) void gemm_wmma_kernel(
    const _Float16* __restrict__ A, const _Float16* __restrict__ Wt,
    const float* __restrict__ bias, float* __restrict__ outf,
    _Float16* __restrict__ outh, int M, int N, int K, int mode) {
  const int lane = threadIdx.x & 31;
  const int wid  = threadIdx.x >> 5;
  const int nbase = blockIdx.x * 128 + (wid >> 2) * 64;
  const int mbase = blockIdx.y * 128 + (wid & 3) * 32;
  if (mbase >= M) return;

  const _Float16* Ar0 = A + (size_t)mbase * K;
  const _Float16* Ar1 = Ar0 + (size_t)16 * K;
  const _Float16* W0  = Wt + (size_t)nbase * K;
  const _Float16* W1  = W0 + (size_t)16 * K;
  const _Float16* W2  = W0 + (size_t)32 * K;
  const _Float16* W3  = W0 + (size_t)48 * K;

  v8f acc0[4] = {};
  v8f acc1[4] = {};

  // ping fragments (kb = 0)
  v16h pa0 = load_a(Ar0, K, lane);
  v16h pa1 = load_a(Ar1, K, lane);
  v16h pb0 = load_b(W0, K, lane);
  v16h pb1 = load_b(W1, K, lane);
  v16h pb2 = load_b(W2, K, lane);
  v16h pb3 = load_b(W3, K, lane);

  for (int kb = 0; kb < K; kb += 64) {
    // issue pong loads (kb+32) before consuming ping
    v16h qa0 = load_a(Ar0 + kb + 32, K, lane);
    v16h qa1 = load_a(Ar1 + kb + 32, K, lane);
    v16h qb0 = load_b(W0 + kb + 32, K, lane);
    v16h qb1 = load_b(W1 + kb + 32, K, lane);
    v16h qb2 = load_b(W2 + kb + 32, K, lane);
    v16h qb3 = load_b(W3 + kb + 32, K, lane);

    acc0[0] = wmma_f16(pa0, pb0, acc0[0]);
    acc1[0] = wmma_f16(pa1, pb0, acc1[0]);
    acc0[1] = wmma_f16(pa0, pb1, acc0[1]);
    acc1[1] = wmma_f16(pa1, pb1, acc1[1]);
    acc0[2] = wmma_f16(pa0, pb2, acc0[2]);
    acc1[2] = wmma_f16(pa1, pb2, acc1[2]);
    acc0[3] = wmma_f16(pa0, pb3, acc0[3]);
    acc1[3] = wmma_f16(pa1, pb3, acc1[3]);

    // issue ping loads for kb+64 before consuming pong
    if (kb + 64 < K) {
      pa0 = load_a(Ar0 + kb + 64, K, lane);
      pa1 = load_a(Ar1 + kb + 64, K, lane);
      pb0 = load_b(W0 + kb + 64, K, lane);
      pb1 = load_b(W1 + kb + 64, K, lane);
      pb2 = load_b(W2 + kb + 64, K, lane);
      pb3 = load_b(W3 + kb + 64, K, lane);
    }

    acc0[0] = wmma_f16(qa0, qb0, acc0[0]);
    acc1[0] = wmma_f16(qa1, qb0, acc1[0]);
    acc0[1] = wmma_f16(qa0, qb1, acc0[1]);
    acc1[1] = wmma_f16(qa1, qb1, acc1[1]);
    acc0[2] = wmma_f16(qa0, qb2, acc0[2]);
    acc1[2] = wmma_f16(qa1, qb2, acc1[2]);
    acc0[3] = wmma_f16(qa0, qb3, acc0[3]);
    acc1[3] = wmma_f16(qa1, qb3, acc1[3]);
  }

  const int rg   = lane >> 4;     // C layout: M = v + 8*rg, N = lane&15
  const int col0 = lane & 15;
#pragma unroll
  for (int mt = 0; mt < 2; ++mt) {
#pragma unroll
    for (int t = 0; t < 4; ++t) {
      int n = nbase + t * 16 + col0;
      float bn = bias[n];
#pragma unroll
      for (int v = 0; v < 8; ++v) {
        int row = mbase + mt * 16 + rg * 8 + v;
        float val = (mt == 0 ? acc0[t][v] : acc1[t][v]) + bn;
        if (mode == 0) {
          if ((n % (3 * cHD)) < cHD) val *= cSCALE;   // pre-scale q
          outh[(size_t)row * N + n] = (_Float16)val;
        } else if (mode == 1) {
          outf[(size_t)row * N + n] = val;
        } else {
          val = val > 0.f ? val : 0.f;
          outh[(size_t)row * N + n] = (_Float16)val;
        }
      }
    }
  }
}

// ---------------- flash attention ----------------
// One wave handles a 16-row query tile of one (b,h); 4 waves per block.
__global__ __launch_bounds__(128) void attn_flash_kernel(
    const _Float16* __restrict__ qkvh, const _Float16* __restrict__ vt,
    const float* __restrict__ lm, _Float16* __restrict__ outh) {
  __shared__ __align__(16) _Float16 Ps[4][16][32];   // per-wave P staging
  const int lane = threadIdx.x & 31;
  const int wid  = threadIdx.x >> 5;
  const int qtiles = cL / 16;
  int gid = blockIdx.x * 4 + wid;
  int b  = gid / (cH * qtiles);
  int rm = gid % (cH * qtiles);
  int h  = rm / qtiles;
  int qt = rm % qtiles;

  const _Float16* qbase = qkvh + ((size_t)b * cL + qt * 16) * c3D + h * (3 * cHD);
  const _Float16* kbase = qkvh + ((size_t)b * cL) * c3D + h * (3 * cHD) + cHD;
  const _Float16* vbase = vt + (size_t)(b * cH + h) * cHD * cL;
  const float* lmb = lm + (size_t)b * cL;

  v16h qa0 = load_a(qbase, c3D, lane);        // hd 0..31
  v16h qa1 = load_a(qbase + 32, c3D, lane);   // hd 32..63

  float mrow[8], lrow[8];
#pragma unroll
  for (int v = 0; v < 8; ++v) { mrow[v] = -1e30f; lrow[v] = 0.f; }
  v8f acc[4] = {};

  const int col = lane & 15;
  const int rg  = lane >> 4;

  for (int kb = 0; kb < cL; kb += 32) {
    // S = q @ k^T for 32 keys (two 16x16 C tiles), K-dim = HD = 64
    v8f s[2] = {};
#pragma unroll
    for (int t = 0; t < 2; ++t) {
      const _Float16* krow = kbase + (size_t)(kb + t * 16) * c3D;
      s[t] = wmma_f16(qa0, load_b(krow, c3D, lane), s[t]);
      s[t] = wmma_f16(qa1, load_b(krow + 32, c3D, lane), s[t]);
    }
    // prefetch V fragments for this key block now: independent of softmax,
    // their latency hides under the VALU-heavy online-softmax update below.
    v16h bv0 = load_b(vbase + (size_t)0 * cL + kb, cL, lane);
    v16h bv1 = load_b(vbase + (size_t)16 * cL + kb, cL, lane);
    v16h bv2 = load_b(vbase + (size_t)32 * cL + kb, cL, lane);
    v16h bv3 = load_b(vbase + (size_t)48 * cL + kb, cL, lane);

    // additive log-mask per key column
#pragma unroll
    for (int t = 0; t < 2; ++t) {
      float lmv = lmb[kb + t * 16 + col];
#pragma unroll
      for (int v = 0; v < 8; ++v) s[t][v] += lmv;
    }
    // online softmax update (row reductions across 16-lane groups)
    float p0[8], p1[8];
#pragma unroll
    for (int v = 0; v < 8; ++v) {
      float mx = fmaxf(s[0][v], s[1][v]);
#pragma unroll
      for (int o = 1; o < 16; o <<= 1) mx = fmaxf(mx, __shfl_xor(mx, o, 16));
      float mnew = fmaxf(mrow[v], mx);
      float sc = __expf(mrow[v] - mnew);
      p0[v] = __expf(s[0][v] - mnew);
      p1[v] = __expf(s[1][v] - mnew);
      float rs = p0[v] + p1[v];
#pragma unroll
      for (int o = 1; o < 16; o <<= 1) rs += __shfl_xor(rs, o, 16);
      lrow[v] = lrow[v] * sc + rs;
      mrow[v] = mnew;
#pragma unroll
      for (int t = 0; t < 4; ++t) acc[t][v] *= sc;
    }
    // stage P (16x32) to LDS in row-major, re-read as A fragment
    __syncthreads();
#pragma unroll
    for (int v = 0; v < 8; ++v) {
      Ps[wid][rg * 8 + v][col]      = (_Float16)p0[v];
      Ps[wid][rg * 8 + v][16 + col] = (_Float16)p1[v];
    }
    __syncthreads();
    v16h pa;
    {
      int m = lane & 15, hf = lane >> 4;
      const _Float16* pp = &Ps[wid][m][hf * 8];
      pa = frag_from_2x8(pp, pp + 16);
    }
    // O += P @ V  (B fragments already in registers)
    acc[0] = wmma_f16(pa, bv0, acc[0]);
    acc[1] = wmma_f16(pa, bv1, acc[1]);
    acc[2] = wmma_f16(pa, bv2, acc[2]);
    acc[3] = wmma_f16(pa, bv3, acc[3]);
  }

  // normalize and write y[b, l, h*64 + hd] (f16)
#pragma unroll
  for (int t = 0; t < 4; ++t) {
#pragma unroll
    for (int v = 0; v < 8; ++v) {
      int row = qt * 16 + rg * 8 + v;
      float ov = acc[t][v] / lrow[v];
      outh[((size_t)b * cL + row) * cD + h * cHD + t * 16 + col] = (_Float16)ov;
    }
  }
}

// ---------------- fused residual + LayerNorm ----------------
// one block per row of D=512; writes fp32 master and f16 copy
__global__ __launch_bounds__(256) void ln_kernel(
    const float* __restrict__ xin, const float* __restrict__ y,
    const float* __restrict__ g, const float* __restrict__ bt,
    float* __restrict__ xout, _Float16* __restrict__ xh) {
  __shared__ float red[256];
  const int row = blockIdx.x;
  const int tid = threadIdx.x;
  const float* xr = xin + (size_t)row * cD;
  const float* yr = y   + (size_t)row * cD;
  float t0 = xr[tid] + yr[tid];
  float t1 = xr[tid + 256] + yr[tid + 256];

  red[tid] = t0 + t1;
  __syncthreads();
  for (int o = 128; o > 0; o >>= 1) {
    if (tid < o) red[tid] += red[tid + o];
    __syncthreads();
  }
  float mu = red[0] * (1.f / cD);
  __syncthreads();
  float d0 = t0 - mu, d1 = t1 - mu;
  red[tid] = d0 * d0 + d1 * d1;
  __syncthreads();
  for (int o = 128; o > 0; o >>= 1) {
    if (tid < o) red[tid] += red[tid + o];
    __syncthreads();
  }
  float rstd = rsqrtf(red[0] * (1.f / cD) + cEPS);
  float o0 = d0 * rstd * g[tid] + bt[tid];
  float o1 = d1 * rstd * g[tid + 256] + bt[tid + 256];
  xout[(size_t)row * cD + tid] = o0;
  xout[(size_t)row * cD + tid + 256] = o1;
  xh[(size_t)row * cD + tid] = (_Float16)o0;
  xh[(size_t)row * cD + tid + 256] = (_Float16)o1;
}

// ---------------- host-side orchestration ----------------
extern "C" void kernel_launch(void* const* d_in, const int* in_sizes, int n_in,
                              void* d_out, int out_size, void* d_ws, size_t ws_size,
                              hipStream_t stream) {
  (void)in_sizes; (void)n_in; (void)out_size; (void)ws_size;
  const float* x     = (const float*)d_in[0];
  const float* mask  = (const float*)d_in[1];
  const float* pos   = (const float*)d_in[2];
  const float* w_qkv = (const float*)d_in[3];
  const float* b_qkv = (const float*)d_in[4];
  const float* w_o   = (const float*)d_in[5];
  const float* b_o   = (const float*)d_in[6];
  const float* w1    = (const float*)d_in[7];
  const float* b1    = (const float*)d_in[8];
  const float* w2    = (const float*)d_in[9];
  const float* b2    = (const float*)d_in[10];
  const float* ln1g  = (const float*)d_in[11];
  const float* ln1b  = (const float*)d_in[12];
  const float* ln2g  = (const float*)d_in[13];
  const float* ln2b  = (const float*)d_in[14];

  char* ws = (char*)d_ws;
  size_t off = 0;
  auto take = [&](size_t bytes) -> char* {
    off = (off + 255) & ~(size_t)255;
    char* p = ws + off;
    off += bytes;
    return p;
  };
  const size_t BLD = (size_t)cBL * cD;          // 4,194,304
  float*    xf32  = (float*)   take(BLD * 4);
  _Float16* xh    = (_Float16*)take(BLD * 2);
  _Float16* qkvh  = (_Float16*)take((size_t)cBL * c3D * 2);
  _Float16* vtb   = (_Float16*)take(BLD * 2);
  _Float16* yh    = (_Float16*)take(BLD * 2);
  float*    y32   = (float*)   take(BLD * 4);
  _Float16* ffh   = (_Float16*)take((size_t)cBL * cFF * 2);
  float*    lmb   = (float*)   take((size_t)cBL * 4);
  _Float16* wqkvt = (_Float16*)take((size_t)cNL * c3D * cD * 2);
  _Float16* wot   = (_Float16*)take((size_t)cNL * cD * cD * 2);
  _Float16* w1t   = (_Float16*)take((size_t)cNL * cFF * cD * 2);
  _Float16* w2t   = (_Float16*)take((size_t)cNL * cD * cFF * 2);

  // prep: x + pos, log(mask), weight convert+transpose
  add_pos_kernel<<<(BLD + 255) / 256, 256, 0, stream>>>(x, pos, xf32, xh);
  log_mask_kernel<<<(cBL + 255) / 256, 256, 0, stream>>>(mask, lmb);
  for (int l = 0; l < cNL; ++l) {
    conv_wt_kernel<<<((size_t)cD * c3D + 255) / 256, 256, 0, stream>>>(
        w_qkv + (size_t)l * cD * c3D, wqkvt + (size_t)l * c3D * cD, cD, c3D);
    conv_wt_kernel<<<((size_t)cD * cD + 255) / 256, 256, 0, stream>>>(
        w_o + (size_t)l * cD * cD, wot + (size_t)l * cD * cD, cD, cD);
    conv_wt_kernel<<<((size_t)cD * cFF + 255) / 256, 256, 0, stream>>>(
        w1 + (size_t)l * cD * cFF, w1t + (size_t)l * cFF * cD, cD, cFF);
    conv_wt_kernel<<<((size_t)cFF * cD + 255) / 256, 256, 0, stream>>>(
        w2 + (size_t)l * cFF * cD, w2t + (size_t)l * cD * cFF, cFF, cD);
  }

  const dim3 gemm_block(256);
  for (int l = 0; l < cNL; ++l) {
    const _Float16* wq = wqkvt + (size_t)l * c3D * cD;
    const _Float16* wo = wot   + (size_t)l * cD * cD;
    const _Float16* wa = w1t   + (size_t)l * cFF * cD;
    const _Float16* wb = w2t   + (size_t)l * cD * cFF;

    // qkv = x @ Wqkv + b  (f16 out, q pre-scaled)
    gemm_wmma_kernel<<<dim3(c3D / 128, cBL / 128), gemm_block, 0, stream>>>(
        xh, wq, b_qkv + (size_t)l * c3D, nullptr, qkvh, cBL, c3D, cD, 0);
    // Vt for P@V fragments
    transpose_v_kernel<<<(BLD + 255) / 256, 256, 0, stream>>>(qkvh, vtb);
    // flash attention
    attn_flash_kernel<<<(cB * cH * (cL / 16)) / 4, 128, 0, stream>>>(qkvh, vtb, lmb, yh);
    // y = attn @ Wo + b_o (f32 out)
    gemm_wmma_kernel<<<dim3(cD / 128, cBL / 128), gemm_block, 0, stream>>>(
        yh, wo, b_o + (size_t)l * cD, y32, nullptr, cBL, cD, cD, 1);
    // x = LN(x + y)
    ln_kernel<<<cBL, 256, 0, stream>>>(xf32, y32, ln1g + (size_t)l * cD,
                                       ln1b + (size_t)l * cD, xf32, xh);
    // ff = relu(x @ W1 + b1) (f16 out)
    gemm_wmma_kernel<<<dim3(cFF / 128, cBL / 128), gemm_block, 0, stream>>>(
        xh, wa, b1 + (size_t)l * cFF, nullptr, ffh, cBL, cFF, cD, 2);
    // m = ff @ W2 + b2 (f32 out)
    gemm_wmma_kernel<<<dim3(cD / 128, cBL / 128), gemm_block, 0, stream>>>(
        ffh, wb, b2 + (size_t)l * cD, y32, nullptr, cBL, cD, cFF, 1);
    // x = LN(x + m); final layer writes d_out directly
    float* lnout = (l == cNL - 1) ? (float*)d_out : xf32;
    ln_kernel<<<cBL, 256, 0, stream>>>(xf32, y32, ln2g + (size_t)l * cD,
                                       ln2b + (size_t)l * cD, lnout, xh);
  }
}
